// LoopScaledDotProductAttention_69922067579341
// MI455X (gfx1250) — compile-verified, matrix operand order checked by reading, not run on hardware
//
#include <hip/hip_runtime.h>
#include <hip/hip_bf16.h>

typedef __attribute__((ext_vector_type(16))) __bf16 v16bf;
typedef __attribute__((ext_vector_type(8)))  __bf16 v8bf;   // ds_*_b128
typedef __attribute__((ext_vector_type(4)))  __bf16 v4bf;   // ds_*_b64
typedef __attribute__((ext_vector_type(8)))  float  v8f;

namespace {
constexpr int kB = 4, kH = 16, kS = 4096, kD = 64, kDV = 64;
constexpr int CHK    = 64;            // chunk length
constexpr int NCHK   = kS / CHK;      // 64 sequential chunks per head
constexpr int NSPLIT = 2;             // split Dv across blocks for occupancy
constexpr int DVS    = kDV / NSPLIT;  // 32 output columns per block
constexpr int LDA    = 72;            // padded stride: 144 B rows, 16B-aligned
constexpr float kEps = 1e-5f;

__device__ __forceinline__ float elu1(float x) {
  return x > 0.f ? x + 1.f : __expf(x);   // elu(x)+1
}

// 16-bit operand fragment from a K-innermost LDS tile ([M][K] for A, [N][K]
// for B^T-style operands). ISA 7.12.2: two contiguous 8-element (16 B) runs
// per lane -> exactly two ds_load_b128.
__device__ __forceinline__ v16bf frag_row(const __bf16* buf, int m0, int k0,
                                          int lane) {
  const int half = lane >> 4;
  const __bf16* p = buf + (m0 + (lane & 15)) * LDA + k0 + half * 8;
  v8bf lo = *(const v8bf*)p;          // K = k0+8h .. +7
  v8bf hi = *(const v8bf*)(p + 16);   // K = k0+16+8h .. +7
  return __builtin_shufflevector(lo, hi, 0, 1, 2, 3, 4, 5, 6, 7,
                                 8, 9, 10, 11, 12, 13, 14, 15);
}
} // namespace

__global__ __launch_bounds__(512)
void linattn_chunked_wmma(const float* __restrict__ qg,
                          const float* __restrict__ kg,
                          const float* __restrict__ vg,
                          float* __restrict__ og) {
  __shared__ __bf16 Qs [CHK][LDA];   // elu+1(q), [t][d]
  __shared__ __bf16 Ks [CHK][LDA];   // elu+1(k), [j][d]
  __shared__ __bf16 KsT[kD ][LDA];   // elu+1(k) transposed, [d][j]
  __shared__ __bf16 AmT[CHK][LDA];   // masked scores A[t][j] (K-innermost)
  __shared__ __bf16 VsT[DVS][LDA];   // V slab transposed, [n][j]
  __shared__ __bf16 SsT[DVS][LDA];   // state snapshot transposed, [n][d]
  __shared__ float  zst[kD];         // running k-sum
  __shared__ float  den[CHK];        // reciprocal denominators

  const int tid  = threadIdx.x;
  const int lane = tid & 31;
  const int wave = tid >> 5;        // 0..15
  const int wm   = wave >> 2;       // 16-row tile index 0..3
  const int wn   = wave & 3;        // 16-col tile index 0..3
  const int half = lane >> 4;

  const int head  = blockIdx.x / NSPLIT;   // b*H + h, 0..63
  const int split = blockIdx.x % NSPLIT;   // Dv slab

  const float* qh = qg + (size_t)head * kS * kD;
  const float* kh = kg + (size_t)head * kS * kD;
  const float* vh = vg + (size_t)head * kS * kDV;
  float*       oh = og + (size_t)head * kS * kDV;

  // Zero state and AmT once (upper-right zero tiles of AmT[t][j] are never rewritten).
  for (int i = tid; i < DVS * LDA; i += 512) ((__bf16*)SsT)[i] = (__bf16)0.f;
  for (int i = tid; i < CHK * LDA; i += 512) ((__bf16*)AmT)[i] = (__bf16)0.f;
  if (tid < kD) zst[tid] = 0.f;

  v8f s_acc = {};   // fp32 state tile S[wm-rows][wn-cols] (waves wn<2)
  __syncthreads();

#pragma unroll 1
  for (int c = 0; c < NCHK; ++c) {
    const int rbase = c * CHK;

    // ---- phase 1: stage chunk (4x4 register blocks -> b64 LDS stores) ----
    {
      const int t  = tid & 255;
      const int br = (t >> 4) * 4;               // row group
      const int bc = (t & 15) * 4;               // col group
      if (tid < 256) {                           // Q tile
        float4 x[4];
#pragma unroll
        for (int i = 0; i < 4; ++i)
          x[i] = *(const float4*)(qh + (size_t)(rbase + br + i) * kD + bc);
#pragma unroll
        for (int i = 0; i < 4; ++i) {
          v4bf r = {(__bf16)elu1(x[i].x), (__bf16)elu1(x[i].y),
                    (__bf16)elu1(x[i].z), (__bf16)elu1(x[i].w)};
          *(v4bf*)&Qs[br + i][bc] = r;
        }
      } else {                                   // K tile: row-major + transpose
        float4 x[4];
#pragma unroll
        for (int i = 0; i < 4; ++i)
          x[i] = *(const float4*)(kh + (size_t)(rbase + br + i) * kD + bc);
        __bf16 kb[4][4];
#pragma unroll
        for (int i = 0; i < 4; ++i) {
          const float* xf = (const float*)&x[i];
#pragma unroll
          for (int j = 0; j < 4; ++j) kb[i][j] = (__bf16)elu1(xf[j]);
        }
#pragma unroll
        for (int i = 0; i < 4; ++i) {
          v4bf r = {kb[i][0], kb[i][1], kb[i][2], kb[i][3]};
          *(v4bf*)&Ks[br + i][bc] = r;
        }
#pragma unroll
        for (int j = 0; j < 4; ++j) {
          v4bf cvec = {kb[0][j], kb[1][j], kb[2][j], kb[3][j]};
          *(v4bf*)&KsT[bc + j][br] = cvec;
        }
      }
      if (tid < 128) {                           // V slab (64x32), transposed
        const int vr = (tid >> 3) * 4;
        const int vc = (tid & 7) * 4;
        float4 x[4];
#pragma unroll
        for (int i = 0; i < 4; ++i)
          x[i] = *(const float4*)(vh + (size_t)(rbase + vr + i) * kDV +
                                  split * DVS + vc);
#pragma unroll
        for (int j = 0; j < 4; ++j) {
          v4bf cvec = {(__bf16)((const float*)&x[0])[j],
                       (__bf16)((const float*)&x[1])[j],
                       (__bf16)((const float*)&x[2])[j],
                       (__bf16)((const float*)&x[3])[j]};
          *(v4bf*)&VsT[vc + j][vr] = cvec;
        }
      }
    }
    __syncthreads();

    // ---- phase 2: O += Qf @ S_prev ; A^T = mask(Kf @ Qf^T) -> AmT ----
    v8f o_acc = {};
    if (wn < 2) {
      v16bf qa0 = frag_row(&Qs[0][0],  wm * 16, 0,  lane);
      v16bf qa1 = frag_row(&Qs[0][0],  wm * 16, 32, lane);
      v16bf sb0 = frag_row(&SsT[0][0], wn * 16, 0,  lane);
      v16bf sb1 = frag_row(&SsT[0][0], wn * 16, 32, lane);
      o_acc = __builtin_amdgcn_wmma_f32_16x16x32_bf16(
          false, qa0, false, sb0, (short)0, o_acc, false, false);
      o_acc = __builtin_amdgcn_wmma_f32_16x16x32_bf16(
          false, qa1, false, sb1, (short)0, o_acc, false, false);
    }
    if (wm <= wn) {   // A^T tile rows j=wm-group, cols t=wn-group; j<=t survives
      v16bf ka0 = frag_row(&Ks[0][0], wm * 16, 0,  lane);
      v16bf ka1 = frag_row(&Ks[0][0], wm * 16, 32, lane);
      v16bf qb0 = frag_row(&Qs[0][0], wn * 16, 0,  lane);   // B = Qs^T
      v16bf qb1 = frag_row(&Qs[0][0], wn * 16, 32, lane);
      v8f t_acc = {};
      t_acc = __builtin_amdgcn_wmma_f32_16x16x32_bf16(
          false, ka0, false, qb0, (short)0, t_acc, false, false);
      t_acc = __builtin_amdgcn_wmma_f32_16x16x32_bf16(
          false, ka1, false, qb1, (short)0, t_acc, false, false);
      const int t  = wn * 16 + (lane & 15);      // query index (C-layout col)
      const int j0 = wm * 16 + half * 8;         // key indices (C-layout rows)
      v8bf st;
#pragma unroll
      for (int v = 0; v < 8; ++v)
        st[v] = (__bf16)(((j0 + v) <= t) ? t_acc[v] : 0.f);
      *(v8bf*)&AmT[t][j0] = st;                  // one ds_store_b128
    }
    __syncthreads();

    // ---- phase 3: O += A @ V ; S += Kf^T @ V ; reciprocal denominators ----
    if (wn < 2) {
      const v16bf vb0 = frag_row(&VsT[0][0], wn * 16, 0,  lane);
      const v16bf vb1 = frag_row(&VsT[0][0], wn * 16, 32, lane);
      v16bf a0 = frag_row(&AmT[0][0], wm * 16, 0,  lane);
      v16bf a1 = frag_row(&AmT[0][0], wm * 16, 32, lane);
      o_acc = __builtin_amdgcn_wmma_f32_16x16x32_bf16(
          false, a0, false, vb0, (short)0, o_acc, false, false);
      o_acc = __builtin_amdgcn_wmma_f32_16x16x32_bf16(
          false, a1, false, vb1, (short)0, o_acc, false, false);
      v16bf kt0 = frag_row(&KsT[0][0], wm * 16, 0,  lane);  // A = Kf^T
      v16bf kt1 = frag_row(&KsT[0][0], wm * 16, 32, lane);
      s_acc = __builtin_amdgcn_wmma_f32_16x16x32_bf16(
          false, kt0, false, vb0, (short)0, s_acc, false, false);
      s_acc = __builtin_amdgcn_wmma_f32_16x16x32_bf16(
          false, kt1, false, vb1, (short)0, s_acc, false, false);
    }
    if (tid < CHK) {   // den[t] = 1 / (rowsum(A[t]) + q_t·z_prev + eps)
      const int t = tid;
      float rs = 0.f, qz = 0.f;
#pragma unroll
      for (int j = 0; j < CHK; j += 8) {
        const v8bf a = *(const v8bf*)&AmT[t][j];
        const v8bf q = *(const v8bf*)&Qs[t][j];
#pragma unroll
        for (int e = 0; e < 8; ++e) {
          rs += (float)a[e];
          qz += (float)q[e] * zst[j + e];
        }
      }
      den[t] = 1.f / (rs + qz + kEps);
    }
    __syncthreads();

    // ---- phase 4: z += colsum(Kf); out = O*rden; refresh state snapshot ----
    if (tid < kD) {
      float zs = 0.f;
#pragma unroll
      for (int r = 0; r < CHK; r += 8) {
        const v8bf kr = *(const v8bf*)&KsT[tid][r];
#pragma unroll
        for (int e = 0; e < 8; ++e) zs += (float)kr[e];
      }
      zst[tid] += zs;
    }
    if (wn < 2) {
      const int col = wn * 16 + (lane & 15);
      const int r0  = wm * 16 + half * 8;
      const v8f rd = *(const v8f*)&den[r0];      // 8 contiguous reciprocals
      v8bf st;
#pragma unroll
      for (int v = 0; v < 8; ++v) {
        oh[(size_t)(rbase + r0 + v) * kDV + split * DVS + col] =
            o_acc[v] * rd[v];
        st[v] = (__bf16)s_acc[v];
      }
      *(v8bf*)&SsT[col][r0] = st;                // one ds_store_b128
    }
    __syncthreads();
  }
}

extern "C" void kernel_launch(void* const* d_in, const int* in_sizes, int n_in,
                              void* d_out, int out_size, void* d_ws, size_t ws_size,
                              hipStream_t stream) {
  (void)in_sizes; (void)n_in; (void)out_size; (void)d_ws; (void)ws_size;
  const float* q = (const float*)d_in[0];
  const float* k = (const float*)d_in[1];
  const float* v = (const float*)d_in[2];
  float* out = (float*)d_out;
  dim3 grid(kB * kH * NSPLIT);   // 128 workgroups
  dim3 block(512);               // 16 wave32s per WGP
  linattn_chunked_wmma<<<grid, block, 0, stream>>>(q, k, v, out);
}
